// MultiHeadVQVAE_47656957116800
// MI455X (gfx1250) — matrix operand order repaired and checked by reading, not compiled, not run on hardware
//
#include <hip/hip_runtime.h>
#include <hip/hip_bf16.h>

// ---------------------------------------------------------------------------
// MultiHeadVQVAE forward for MI455X (gfx1250, wave32, WMMA).
//  - GEMMs: bf16 WMMA 16x16x32 / f32 accum; f32->bf16 RNE conversion fused
//    into LDS staging; the 2x-bf16 pack is a single v_perm_b32. Double-
//    buffered LDS so global fetch overlaps the WMMA pipe (one barrier/k-step).
//  - Sinkhorn: log-domain (u,v) potentials, Q never materialized. Cost matrix
//    is pre-scaled once and also stored transposed so BOTH the row pass and
//    the column pass stream coalesced, L2-resident data (192 MB L2).
//  - Assignment: argmax_j (s_ij + v_j) == argmax_j Q_ij (row scaling cancels).
// ---------------------------------------------------------------------------

typedef __attribute__((ext_vector_type(16))) __bf16 v16bf;
typedef __attribute__((ext_vector_type(8)))  float  v8f;

union FragAB { uint4 q[2]; v16bf v; };

#define NB      32768L           // batch rows
#define NE      256              // codes per head
#define EDIM    64
#define NHEADS  4
#define HSZ     (NB * NE)        // per-head cost matrix elems
#define LOG_B   10.397207708399179f   // log(32768)
#define LOG_K   5.545177444479562f    // log(256)

// ---- workspace layout (float indices) -------------------------------------
#define WS_H1   0L               // 32768x512
#define WS_H2   16777216L        // 32768x256
#define WS_LAT  25165824L        // 32768x256
#define WS_D    33554432L        // 4 x 32768x256  (raw d, then scaled s)
#define WS_DT   67108864L        // 4 x 256x32768  (s transposed)
#define WS_XQ   100663296L       // 32768x256
#define WS_G1   109051904L       // 32768x256
#define WS_G2   117440512L       // 32768x512
#define WS_U    134217728L       // 4 x 32768
#define WS_RN   134348800L       // 4 x 32768
#define WS_V    134479872L       // 4 x 256
#define WS_CN   134480896L       // 4 x 256
#define WS_ST   134481920L       // 4 x {middle, 1/(amp*eps)}
#define WS_MM   134481928L       // 4 x {maxkey, minkey} (uint)
#define WS_LS   134481936L       // loss accumulator
#define WS_CT   134482176L       // 4x256 int counts

// ---- d_out layout ----------------------------------------------------------
#define OUT_LOSS  25165824L      // after 32768*768 'out'
#define OUT_IDX   25165825L      // 32768*4
#define OUT_UNUSE 25296897L

__device__ __forceinline__ unsigned int bfrne(unsigned int x) { // round bits
  return x + 0x7FFFu + ((x >> 16) & 1u);
}
__device__ __forceinline__ unsigned short f2bf(float f) {   // RNE f32->bf16
  return (unsigned short)(bfrne(__float_as_uint(f)) >> 16);
}
// two f32 -> packed 2x bf16 (RNE). Final pack is ONE v_perm_b32: selector
// 0x07060302 takes bytes{7,6}=hi16(xb) and bytes{3,2}=hi16(xa).
__device__ __forceinline__ unsigned int pack2(float a, float b) {
  unsigned int xa = bfrne(__float_as_uint(a));
  unsigned int xb = bfrne(__float_as_uint(b));
  return __builtin_amdgcn_perm(xb, xa, 0x07060302u);
}
// order-preserving float<->uint keys so atomicMax/Min work on floats
__device__ __forceinline__ unsigned int fkey(float f) {
  unsigned int b = __float_as_uint(f);
  return (b & 0x80000000u) ? ~b : (b | 0x80000000u);
}
__device__ __forceinline__ float unkey(unsigned int k) {
  unsigned int b = (k & 0x80000000u) ? (k & 0x7FFFFFFFu) : ~k;
  return __uint_as_float(b);
}

// ---------------------------------------------------------------------------
// Tiled GEMM, 128x64 block tile, BK=32, 8 waves (4x2), each wave 32x32 via
// 2x2 WMMA fragments. Double-buffered LDS staging (one barrier per k-step).
// EPI: 0=bias+relu, 1=bias, 2=distance epilogue (rn[m]+cn[n]-2*acc with
// fused global min/max). BT: B stored NxK (codebooks).
// ---------------------------------------------------------------------------
template<int EPI, bool BT>
__global__ __launch_bounds__(256)
void gemm_wmma(const float* __restrict__ A, int lda, long aHeadOff,
               const float* __restrict__ Bm, int ldb, long bHeadOff,
               const float* __restrict__ bias,
               float* __restrict__ C, int ldc, long cHeadOff,
               int M, int N, int K,
               const float* __restrict__ rn, const float* __restrict__ cn,
               unsigned int* __restrict__ minmax)
{
  __shared__ __align__(16) unsigned short As[2][128 * 40];  // [row][k] pad 40
  __shared__ __align__(16) unsigned short Bs[2][64 * 40];   // [n][k]   pad 40

  const int z = blockIdx.z;
  A  += (long)z * aHeadOff;
  Bm += (long)z * bHeadOff;
  C  += (long)z * cHeadOff;
  if (EPI == 2) { rn += (long)z * M; cn += (long)z * N; }

  const int tid  = threadIdx.x;
  const int lane = tid & 31, wid = tid >> 5;
  const int wm = wid >> 1, wn = wid & 1;
  const int bm = blockIdx.x * 128, bn = blockIdx.y * 64;

  const float z8 = 0.0f;
  v8f c00 = {z8,z8,z8,z8,z8,z8,z8,z8};
  v8f c01 = c00, c10 = c00, c11 = c00;

  const int r0  = tid >> 3, ca4 = (tid & 7) * 4;    // A loader: 128x8 float4
  const int kb0 = tid >> 4, nb4 = (tid & 15) * 4;   // B loader (row-major KxN)
  const int nb0 = tid >> 3, kb4 = (tid & 7) * 4;    // B loader (NxK, transposed)

  const int KT = K / 32;
  float4 ra[4], rb[2];

  auto fetchA = [&](int kt) {
    const int k0 = kt * 32;
    #pragma unroll
    for (int it = 0; it < 4; ++it)
      ra[it] = *reinterpret_cast<const float4*>(A + (long)(bm + r0 + it * 32) * lda + k0 + ca4);
  };
  auto storeA = [&](int buf) {
    #pragma unroll
    for (int it = 0; it < 4; ++it)
      *reinterpret_cast<uint2*>(&As[buf][(r0 + it * 32) * 40 + ca4]) =
          make_uint2(pack2(ra[it].x, ra[it].y), pack2(ra[it].z, ra[it].w));
  };
  auto fetchB = [&](int kt) {
    const int k0 = kt * 32;
    if (BT) {
      #pragma unroll
      for (int it = 0; it < 2; ++it)
        rb[it] = *reinterpret_cast<const float4*>(Bm + (long)(bn + nb0 + it * 32) * ldb + k0 + kb4);
    } else {
      #pragma unroll
      for (int it = 0; it < 2; ++it)
        rb[it] = *reinterpret_cast<const float4*>(Bm + (long)(k0 + kb0 + it * 16) * ldb + bn + nb4);
    }
  };
  auto storeB = [&](int buf) {
    if (BT) {
      #pragma unroll
      for (int it = 0; it < 2; ++it)
        *reinterpret_cast<uint2*>(&Bs[buf][(nb0 + it * 32) * 40 + kb4]) =
            make_uint2(pack2(rb[it].x, rb[it].y), pack2(rb[it].z, rb[it].w));
    } else {
      #pragma unroll
      for (int it = 0; it < 2; ++it) {
        const int kk = kb0 + it * 16;
        Bs[buf][(nb4 + 0) * 40 + kk] = f2bf(rb[it].x);
        Bs[buf][(nb4 + 1) * 40 + kk] = f2bf(rb[it].y);
        Bs[buf][(nb4 + 2) * 40 + kk] = f2bf(rb[it].z);
        Bs[buf][(nb4 + 3) * 40 + kk] = f2bf(rb[it].w);
      }
    }
  };

  fetchA(0); fetchB(0);
  storeA(0); storeB(0);
  __syncthreads();

  for (int kt = 0; kt < KT; ++kt) {
    const int  cur  = kt & 1;
    const bool more = (kt + 1) < KT;
    if (more) { fetchA(kt + 1); fetchB(kt + 1); }   // overlap with WMMA below

    const unsigned short* pA = As[cur];
    const unsigned short* pB = Bs[cur];

    // fragments per ISA 7.12.2 lane layouts
    FragAB a0, a1, b0, b1;
    {
      int row = wm * 32 + (lane & 15);
      int kb  = (lane < 16) ? 0 : 8;            // A: lanes0-15 K{0..7,16..23}
      a0.q[0] = *reinterpret_cast<const uint4*>(&pA[row * 40 + kb]);
      a0.q[1] = *reinterpret_cast<const uint4*>(&pA[row * 40 + kb + 16]);
      row += 16;
      a1.q[0] = *reinterpret_cast<const uint4*>(&pA[row * 40 + kb]);
      a1.q[1] = *reinterpret_cast<const uint4*>(&pA[row * 40 + kb + 16]);
    }
    {
      int col = wn * 32 + (lane & 15);
      int kb  = (lane < 16) ? 0 : 8;            // B: lanes0-15 K{0..7,16..23}
      b0.q[0] = *reinterpret_cast<const uint4*>(&pB[col * 40 + kb]);
      b0.q[1] = *reinterpret_cast<const uint4*>(&pB[col * 40 + kb + 16]);
      col += 16;
      b1.q[0] = *reinterpret_cast<const uint4*>(&pB[col * 40 + kb]);
      b1.q[1] = *reinterpret_cast<const uint4*>(&pB[col * 40 + kb + 16]);
    }

    c00 = __builtin_amdgcn_wmma_f32_16x16x32_bf16(false, a0.v, false, b0.v, (short)0, c00, false, false);
    c01 = __builtin_amdgcn_wmma_f32_16x16x32_bf16(false, a0.v, false, b1.v, (short)0, c01, false, false);
    c10 = __builtin_amdgcn_wmma_f32_16x16x32_bf16(false, a1.v, false, b0.v, (short)0, c10, false, false);
    c11 = __builtin_amdgcn_wmma_f32_16x16x32_bf16(false, a1.v, false, b1.v, (short)0, c11, false, false);

    if (more) { storeA(cur ^ 1); storeB(cur ^ 1); }
    __syncthreads();
  }

  // epilogue. C/D layout: lanes0-15 N=lane, M=vgpr; lanes16-31 M=8+vgpr
  float lmin = 3.4e38f, lmax = -3.4e38f;
  const int nlo = lane & 15;
  const int mhi = (lane < 16) ? 0 : 8;
  #pragma unroll
  for (int f = 0; f < 4; ++f) {
    v8f acc = (f == 0) ? c00 : (f == 1) ? c01 : (f == 2) ? c10 : c11;
    int gm = bm + wm * 32 + (f >> 1) * 16 + mhi;
    int gn = bn + wn * 32 + (f & 1) * 16 + nlo;
    float cnv = 0.f, bv = 0.f;
    if (EPI == 2) cnv = cn[gn]; else bv = bias[gn];
    #pragma unroll
    for (int r = 0; r < 8; ++r) {
      float o;
      if (EPI == 0)      { o = acc[r] + bv; o = o > 0.f ? o : 0.f; }
      else if (EPI == 1) { o = acc[r] + bv; }
      else {
        o = rn[gm + r] + cnv - 2.f * acc[r];
        lmin = fminf(lmin, o); lmax = fmaxf(lmax, o);
      }
      C[(long)(gm + r) * ldc + gn] = o;
    }
  }
  if (EPI == 2) {
    for (int m = 16; m >= 1; m >>= 1) {
      lmax = fmaxf(lmax, __shfl_xor(lmax, m, 32));
      lmin = fminf(lmin, __shfl_xor(lmin, m, 32));
    }
    if (lane == 0) {
      atomicMax(&minmax[z * 2 + 0], fkey(lmax));
      atomicMin(&minmax[z * 2 + 1], fkey(lmin));
    }
  }
}

// ---- squared row norms (rows of length 64), one wave per row ---------------
__global__ __launch_bounds__(256)
void sqnorm_kernel(const float* __restrict__ src, float* __restrict__ out,
                   int rowStride, long headStride, int rowsPerHead)
{
  int head = blockIdx.y;
  int row  = blockIdx.x * 8 + (threadIdx.x >> 5);
  if (row >= rowsPerHead) return;
  int lane = threadIdx.x & 31;
  const float2 v = *reinterpret_cast<const float2*>(
      src + (long)head * headStride + (long)row * rowStride + lane * 2);
  float s = v.x * v.x + v.y * v.y;
  for (int m = 16; m >= 1; m >>= 1) s += __shfl_xor(s, m, 32);
  if (lane == 0) out[(long)head * rowsPerHead + row] = s;
}

// ---- per-head normalization constants from fused min/max -------------------
__global__ void head_stats_kernel(const unsigned int* __restrict__ mm,
                                  float* __restrict__ stats)
{
  int h = threadIdx.x;
  if (h < NHEADS) {
    float mx = unkey(mm[h * 2 + 0]);
    float mn = unkey(mm[h * 2 + 1]);
    float middle = 0.5f * (mx + mn);
    float amp    = mx - middle + 1e-5f;
    stats[h * 2 + 0] = middle;
    stats[h * 2 + 1] = 1.0f / (amp * 0.003f);   // s_ij = (middle - d_ij) * this
  }
}

// ---- one-time: D <- s = (middle-d)*inv (in place) and DT <- s^T ------------
// 64x64 LDS tile; makes the 100x column passes fully coalesced.
__global__ __launch_bounds__(256)
void scale_transpose(float* __restrict__ D, float* __restrict__ DT,
                     const float* __restrict__ stats)
{
  __shared__ float tile[64][65];
  const int head = blockIdx.z;
  const long base = (long)head * HSZ;
  const int m0 = blockIdx.x * 64, n0 = blockIdx.y * 64;
  const float middle = stats[head * 2], inv = stats[head * 2 + 1];
  const int tx = threadIdx.x & 15, ty = threadIdx.x >> 4;     // 16x16 threads
  #pragma unroll
  for (int i = 0; i < 4; ++i) {
    const int mm = m0 + ty + 16 * i;
    float* p = D + base + (long)mm * NE + n0 + tx * 4;
    const float4 v = *reinterpret_cast<const float4*>(p);
    const float4 s = make_float4((middle - v.x) * inv, (middle - v.y) * inv,
                                 (middle - v.z) * inv, (middle - v.w) * inv);
    tile[ty + 16 * i][tx * 4 + 0] = s.x;
    tile[ty + 16 * i][tx * 4 + 1] = s.y;
    tile[ty + 16 * i][tx * 4 + 2] = s.z;
    tile[ty + 16 * i][tx * 4 + 3] = s.w;
    *reinterpret_cast<float4*>(p) = s;
  }
  __syncthreads();
  #pragma unroll
  for (int i = 0; i < 4; ++i) {
    const int nn = n0 + ty + 16 * i;
    const float4 o = make_float4(tile[tx * 4 + 0][ty + 16 * i],
                                 tile[tx * 4 + 1][ty + 16 * i],
                                 tile[tx * 4 + 2][ty + 16 * i],
                                 tile[tx * 4 + 3][ty + 16 * i]);
    *reinterpret_cast<float4*>(DT + base + (long)nn * NB + m0 + tx * 4) = o;
  }
}

// ---- Sinkhorn row update: u_i = -logB - LSE_j(s_ij + v_j), wave per row ----
__global__ __launch_bounds__(256)
void sinkhorn_row(const float* __restrict__ S, const float* __restrict__ v,
                  float* __restrict__ u)
{
  int head = blockIdx.y;
  int row  = blockIdx.x * 8 + (threadIdx.x >> 5);
  int lane = threadIdx.x & 31;
  const float* srow = S + (long)head * HSZ + (long)row * NE;
  const float* vh   = v + head * NE;
  float s[8]; float m = -3.4e38f;
  #pragma unroll
  for (int t = 0; t < 8; ++t) {
    int j = lane + t * 32;
    s[t] = srow[j] + vh[j];
    m = fmaxf(m, s[t]);
  }
  for (int k = 16; k >= 1; k >>= 1) m = fmaxf(m, __shfl_xor(m, k, 32));
  float sum = 0.f;
  #pragma unroll
  for (int t = 0; t < 8; ++t) sum += __expf(s[t] - m);
  for (int k = 16; k >= 1; k >>= 1) sum += __shfl_xor(sum, k, 32);
  if (lane == 0) u[(long)head * NB + row] = -(__logf(sum) + m) - LOG_B;
}

// ---- Sinkhorn col update on transposed S: coalesced streaming --------------
__global__ __launch_bounds__(256)
void sinkhorn_col(const float* __restrict__ ST, const float* __restrict__ u,
                  float* __restrict__ v)
{
  __shared__ float red[8];
  int head = blockIdx.y, j = blockIdx.x;
  int tid = threadIdx.x, lane = tid & 31, wid = tid >> 5;
  const float* srow = ST + (long)head * HSZ + (long)j * NB;   // contiguous!
  const float* uh   = u + (long)head * NB;

  float m = -3.4e38f;
  for (int i = tid; i < NB; i += 256) {
    if (i + 4096 < NB) __builtin_prefetch(&srow[i + 4096], 0, 0);
    m = fmaxf(m, srow[i] + uh[i]);
  }
  for (int k = 16; k >= 1; k >>= 1) m = fmaxf(m, __shfl_xor(m, k, 32));
  if (lane == 0) red[wid] = m;
  __syncthreads();
  m = fmaxf(fmaxf(fmaxf(red[0], red[1]), fmaxf(red[2], red[3])),
            fmaxf(fmaxf(red[4], red[5]), fmaxf(red[6], red[7])));
  __syncthreads();
  float sum = 0.f;
  for (int i = tid; i < NB; i += 256)
    sum += __expf(srow[i] + uh[i] - m);
  for (int k = 16; k >= 1; k >>= 1) sum += __shfl_xor(sum, k, 32);
  if (lane == 0) red[wid] = sum;
  __syncthreads();
  if (tid == 0) {
    float tot = red[0] + red[1] + red[2] + red[3] + red[4] + red[5] + red[6] + red[7];
    v[head * NE + j] = -(__logf(tot) + m) - LOG_K;
  }
}

// ---- fused argmax + codebook gather + losses + counts + index output -------
__global__ __launch_bounds__(256)
void vq_assign(const float* __restrict__ S, const float* __restrict__ v,
               const float* __restrict__ lat, const float* __restrict__ cb,
               float* __restrict__ xq, float* __restrict__ out_idx,
               float* __restrict__ lossAcc, int* __restrict__ counts)
{
  int head = blockIdx.y;
  int row  = blockIdx.x * 8 + (threadIdx.x >> 5);
  int lane = threadIdx.x & 31;
  const float* srow = S + (long)head * HSZ + (long)row * NE;
  const float* vh   = v + head * NE;

  float bv = -3.4e38f; int bi = 0x7fffffff;
  #pragma unroll
  for (int t = 0; t < 8; ++t) {
    int j = lane + t * 32;
    float s = srow[j] + vh[j];
    if (s > bv || (s == bv && j < bi)) { bv = s; bi = j; }
  }
  for (int k = 16; k >= 1; k >>= 1) {
    float ov = __shfl_xor(bv, k, 32);
    int   oi = __shfl_xor(bi, k, 32);
    if (ov > bv || (ov == bv && oi < bi)) { bv = ov; bi = oi; }
  }
  // gather e = emb[idx] (64 floats, 2 per lane), loss, scatter
  const float2 ev = *reinterpret_cast<const float2*>(
      cb + (long)head * (NE * EDIM) + (long)bi * EDIM + lane * 2);
  const float2 lv = *reinterpret_cast<const float2*>(
      lat + (long)row * 256 + head * EDIM + lane * 2);
  float d0 = ev.x - lv.x, d1 = ev.y - lv.y;
  float ls = d0 * d0 + d1 * d1;
  for (int k = 16; k >= 1; k >>= 1) ls += __shfl_xor(ls, k, 32);
  *reinterpret_cast<float2*>(xq + (long)row * 256 + head * EDIM + lane * 2) = ev;
  if (lane == 0) {
    atomicAdd(lossAcc, ls);
    atomicAdd(&counts[head * NE + bi], 1);
    out_idx[(long)row * NHEADS + head] = (float)bi;
  }
}

__global__ void init_kernel(float* __restrict__ v, unsigned int* __restrict__ mm,
                            float* __restrict__ lossAcc, int* __restrict__ counts)
{
  int t = threadIdx.x;
  if (t < 1024) { v[t] = 0.f; counts[t] = 0; }
  if (t < NHEADS) { mm[t * 2] = 0u; mm[t * 2 + 1] = 0xFFFFFFFFu; }
  if (t == 0) *lossAcc = 0.f;
}

__global__ __launch_bounds__(1024)
void finalize_kernel(const int* __restrict__ counts, const float* __restrict__ lossAcc,
                     float* __restrict__ outLoss, float* __restrict__ outUnused)
{
  __shared__ int red[32];
  int t = threadIdx.x;
  int c = (counts[t] == 0) ? 1 : 0;
  for (int k = 16; k >= 1; k >>= 1) c += __shfl_xor(c, k, 32);
  if ((t & 31) == 0) red[t >> 5] = c;
  __syncthreads();
  if (t == 0) {
    int s = 0;
    for (int i = 0; i < 32; ++i) s += red[i];
    *outUnused = (float)s;
    // vq_loss = mean_h[(1+BETA)*mse_h] = (1.25/(4*B*64)) * sum diff^2
    *outLoss = *lossAcc * (1.25f / (4.0f * 32768.0f * 64.0f));
  }
}

// ---------------------------------------------------------------------------
extern "C" void kernel_launch(void* const* d_in, const int* in_sizes, int n_in,
                              void* d_out, int out_size, void* d_ws, size_t ws_size,
                              hipStream_t stream)
{
  const float* x   = (const float*)d_in[0];
  const float* ew0 = (const float*)d_in[1];  const float* eb0 = (const float*)d_in[2];
  const float* ew1 = (const float*)d_in[3];  const float* eb1 = (const float*)d_in[4];
  const float* ew2 = (const float*)d_in[5];  const float* eb2 = (const float*)d_in[6];
  const float* dw0 = (const float*)d_in[7];  const float* db0 = (const float*)d_in[8];
  const float* dw1 = (const float*)d_in[9];  const float* db1 = (const float*)d_in[10];
  const float* dw2 = (const float*)d_in[11]; const float* db2 = (const float*)d_in[12];
  const float* cb  = (const float*)d_in[13];
  float* out = (float*)d_out;

  float* ws = (float*)d_ws;
  float* h1  = ws + WS_H1;   float* h2  = ws + WS_H2;
  float* lat = ws + WS_LAT;  float* Db  = ws + WS_D;   float* Dt = ws + WS_DT;
  float* xq  = ws + WS_XQ;   float* g1  = ws + WS_G1;  float* g2 = ws + WS_G2;
  float* uu  = ws + WS_U;    float* rn  = ws + WS_RN;
  float* vv  = ws + WS_V;    float* cn  = ws + WS_CN;  float* st = ws + WS_ST;
  unsigned int* mm = (unsigned int*)(ws + WS_MM);
  float* ls  = ws + WS_LS;
  int* counts = (int*)(ws + WS_CT);

  dim3 blk(256);
  init_kernel<<<1, 1024, 0, stream>>>(vv, mm, ls, counts);

  // encoder: 768 -> 512 -> 256 -> 256 (last layer linear)
  gemm_wmma<0,false><<<dim3(256, 8, 1), blk, 0, stream>>>(x,  768, 0, ew0, 512, 0, eb0, h1, 512, 0, 32768, 512, 768, nullptr, nullptr, nullptr);
  gemm_wmma<0,false><<<dim3(256, 4, 1), blk, 0, stream>>>(h1, 512, 0, ew1, 256, 0, eb1, h2, 256, 0, 32768, 256, 512, nullptr, nullptr, nullptr);
  gemm_wmma<1,false><<<dim3(256, 4, 1), blk, 0, stream>>>(h2, 256, 0, ew2, 256, 0, eb2, lat,256, 0, 32768, 256, 256, nullptr, nullptr, nullptr);

  // squared norms: latent rows (per head) and codebook rows
  sqnorm_kernel<<<dim3(4096, NHEADS), blk, 0, stream>>>(lat, rn, 256, 64,    32768);
  sqnorm_kernel<<<dim3(32,   NHEADS), blk, 0, stream>>>(cb,  cn, 64,  16384, 256);

  // batched distance GEMM: d = |lat|^2 + |emb|^2 - 2 lat@emb^T, fused min/max
  gemm_wmma<2,true><<<dim3(256, 4, NHEADS), blk, 0, stream>>>(
      lat, 256, 64, cb, 64, (long)NE * EDIM, nullptr,
      Db, 256, HSZ, 32768, 256, 64, rn, cn, mm);
  head_stats_kernel<<<1, 32, 0, stream>>>(mm, st);

  // one-time pre-scale (D <- s, in place) + transposed copy for the col pass
  scale_transpose<<<dim3(512, 4, NHEADS), blk, 0, stream>>>(Db, Dt, st);

  // 100 log-domain Sinkhorn iterations; S and S^T stay L2-resident
  for (int it = 0; it < 100; ++it) {
    sinkhorn_row<<<dim3(4096, NHEADS), blk, 0, stream>>>(Db, vv, uu);
    sinkhorn_col<<<dim3(256,  NHEADS), blk, 0, stream>>>(Dt, uu, vv);
  }

  // assignment + quantize + losses + counts + indices
  vq_assign<<<dim3(4096, NHEADS), blk, 0, stream>>>(Db, vv, lat, cb, xq,
                                                    out + OUT_IDX, ls, counts);

  // decoder: 256 -> 256 -> 512 -> 768 (last layer linear, writes d_out)
  gemm_wmma<0,false><<<dim3(256, 4, 1),  blk, 0, stream>>>(xq, 256, 0, dw0, 256, 0, db0, g1, 256, 0, 32768, 256, 256, nullptr, nullptr, nullptr);
  gemm_wmma<0,false><<<dim3(256, 8, 1),  blk, 0, stream>>>(g1, 256, 0, dw1, 512, 0, db1, g2, 512, 0, 32768, 512, 256, nullptr, nullptr, nullptr);
  gemm_wmma<1,false><<<dim3(256, 12, 1), blk, 0, stream>>>(g2, 512, 0, dw2, 768, 0, db2, out,768, 0, 32768, 768, 512, nullptr, nullptr, nullptr);

  finalize_kernel<<<1, 1024, 0, stream>>>(counts, ls, out + OUT_LOSS, out + OUT_UNUSE);

  (void)in_sizes; (void)n_in; (void)out_size; (void)ws_size;
}
